// Attention_16922171146838
// MI455X (gfx1250) — matrix operand order repaired and checked by reading, not compiled, not run on hardware
//
#include <hip/hip_runtime.h>
#include <stdint.h>

// ---------------------------------------------------------------------------
// Attention (B=4, S=2048, D=1024), f16 WMMA pipeline for gfx1250 (MI455X).
//   q = X Wq ; k = X Wk ; scores = q k^T / sqrt(64) ; attn = softmax(scores)
//   out = attn X            (V = raw inputs per reference)
// Round 3: depth-4 B-fragment software pipeline + sched_group_barrier hints
// so v_wmma issue overlaps global_load latency; vectorized softmax LDS I/O.
// ---------------------------------------------------------------------------

typedef __attribute__((ext_vector_type(16))) _Float16 v16h;
typedef __attribute__((ext_vector_type(8)))  _Float16 v8h;
typedef __attribute__((ext_vector_type(8)))  float    v8f;

#define B_ 4
#define S_ 2048
#define D_ 1024

// Scheduler interleave hints (no-op if builtin unavailable).
// mask: 0x008 = MFMA/WMMA, 0x020 = VMEM read, 0x100 = DS read
#if __has_builtin(__builtin_amdgcn_sched_group_barrier)
#define SCHED_GROUP(mask, size, id) __builtin_amdgcn_sched_group_barrier(mask, size, id)
#else
#define SCHED_GROUP(mask, size, id)
#endif

static __device__ __forceinline__ v16h cat8(v8h lo, v8h hi) {
    return __builtin_shufflevector(lo, hi, 0,1,2,3,4,5,6,7,8,9,10,11,12,13,14,15);
}

static __device__ __forceinline__ v8f wmma16x16x32(v16h a, v16h b, v8f c) {
    // D = A(16x32) * B(32x16) + C, f32 accumulate
    return __builtin_amdgcn_wmma_f32_16x16x32_f16(false, a, false, b,
                                                  (short)0, c, false, false);
}

// A-fragment (16x32 f16) loader: row base + K-offset, contiguous k = 16*half + s (B loads match)
static __device__ __forceinline__ v16h load_afrag(const _Float16* row, int k0, int half) {
    v8h lo = *(const v8h*)(row + k0 + half * 16);
    v8h hi = *(const v8h*)(row + k0 + half * 16 + 8);
    return cat8(lo, hi);
}

// ---------------------------------------------------------------------------
// Kernel 1: inputs f32 -> f16, row-major Xh [B*S][D] and transposed XhT [B][D][S]
// ---------------------------------------------------------------------------
__global__ __launch_bounds__(256) void cvt_x(const float* __restrict__ X,
                                             _Float16* __restrict__ Xh,
                                             _Float16* __restrict__ XhT) {
    size_t i = (size_t)blockIdx.x * 256 + threadIdx.x;   // < B*S*D
    float v = X[i];
    _Float16 h = (_Float16)v;
    Xh[i] = h;
    int d  = (int)(i % D_);
    size_t bs = i / D_;
    int s  = (int)(bs % S_);
    int b  = (int)(bs / S_);
    XhT[((size_t)b * D_ + d) * S_ + s] = h;
}

// ---------------------------------------------------------------------------
// Kernel 2: weights f32 -> f16 transposed: WT[n][k] = W[k][n]
// ---------------------------------------------------------------------------
__global__ __launch_bounds__(256) void cvt_w(const float* __restrict__ Wq,
                                             const float* __restrict__ Wk,
                                             _Float16* __restrict__ WqT,
                                             _Float16* __restrict__ WkT) {
    size_t i = (size_t)blockIdx.x * 256 + threadIdx.x;   // < D*D
    int n = (int)(i % D_);
    int k = (int)(i / D_);
    WqT[(size_t)n * D_ + k] = (_Float16)Wq[i];
    WkT[(size_t)n * D_ + k] = (_Float16)Wk[i];
}

// ---------------------------------------------------------------------------
// Kernel 3: projections Q = X Wq, K = X Wk   (M=8192, N=1024, K=1024)
// One wave -> 16x128 tile.  B fragments run 4 deep ahead of their WMMA; the
// next A-frag is prefetched per K-step.
// ---------------------------------------------------------------------------
__global__ __launch_bounds__(256) void proj(const _Float16* __restrict__ Xh,
                                            const _Float16* __restrict__ WqT,
                                            const _Float16* __restrict__ WkT,
                                            _Float16* __restrict__ Qh,
                                            _Float16* __restrict__ Kh) {
    int wave  = blockIdx.x * 8 + (threadIdx.x >> 5);
    int which = wave >> 12;          // 0 -> Q, 1 -> K   (4096 wave-tiles each)
    int rem   = wave & 4095;
    int m0    = (rem >> 3) << 4;     // 512 M-tiles of 16
    int n0    = (rem & 7) << 7;      // 8 N-tiles of 128
    int lane  = threadIdx.x & 31;
    int half  = lane >> 4;
    int lm    = lane & 15;

    const _Float16* WT  = which ? WkT : WqT;
    _Float16*       Out = which ? Kh  : Qh;

    v8f acc[8] = {};
    const _Float16* arow = Xh + (size_t)(m0 + lm) * D_;
    const _Float16* brow = WT + (size_t)(n0 + lm) * D_ + half * 16;

    v16h a_cur = load_afrag(arow, 0, half);
    v16h b0 = *(const v16h*)(brow + (size_t)0 * 16 * D_);
    v16h b1 = *(const v16h*)(brow + (size_t)1 * 16 * D_);
    v16h b2 = *(const v16h*)(brow + (size_t)2 * 16 * D_);
    v16h b3 = *(const v16h*)(brow + (size_t)3 * 16 * D_);

    for (int k0 = 0; k0 < D_; k0 += 32) {
        int k1 = (k0 + 32 < D_) ? (k0 + 32) : k0;   // clamp (redundant last)
        v16h a_nxt = load_afrag(arow, k1, half);
#pragma unroll
        for (int j = 0; j < 8; ++j) {
            int jn = j + 4;
            int kn = (jn >= 8) ? k1 : k0;
            jn &= 7;
            v16h bnew = *(const v16h*)(brow + (size_t)jn * 16 * D_ + kn);
            acc[j] = wmma16x16x32(a_cur, b0, acc[j]);
            b0 = b1; b1 = b2; b2 = b3; b3 = bnew;
        }
        a_cur = a_nxt;
        SCHED_GROUP(0x020, 2, 0);                       // A-frag loads first
#pragma unroll
        for (int j = 0; j < 8; ++j) {                   // (2 loads, 1 wmma) x8
            SCHED_GROUP(0x020, 2, 0);
            SCHED_GROUP(0x008, 1, 0);
        }
    }
#pragma unroll
    for (int j = 0; j < 8; ++j)
#pragma unroll
        for (int r = 0; r < 8; ++r)
            Out[(size_t)(m0 + r + 8 * half) * D_ + n0 + j * 16 + lm] =
                (_Float16)acc[j][r];
}

// ---------------------------------------------------------------------------
// Kernel 4: fused scores + softmax + P*V for one 16-query block per workgroup.
//   Phase 1: wave w -> scores[16 x 256]; K-outer, 16 key-tile accumulators,
//            B fragments pipelined 4 deep
//   Phase 2: row softmax (16 rows x 2048), 16 threads/row, v8h LDS I/O
//   Phase 3: wave w -> O[16 x 128] = P(16x2048) * V slice, same pipelining
// Dynamic LDS: Qtile 32KB + Sbuf 64KB + red 1KB + rmax/rsum 128B
// ---------------------------------------------------------------------------
#define SM_QT   0
#define SM_SB   (32768)
#define SM_RED  (32768 + 65536)
#define SM_RMAX (32768 + 65536 + 1024)
#define SM_RSUM (32768 + 65536 + 1024 + 64)
#define SMEM_BYTES (32768 + 65536 + 1024 + 64 + 64)

__global__ __launch_bounds__(256) void attn(const _Float16* __restrict__ Qh,
                                            const _Float16* __restrict__ Kh,
                                            const _Float16* __restrict__ XhT,
                                            float* __restrict__ out) {
    extern __shared__ char smem[];
    _Float16* Qt   = (_Float16*)(smem + SM_QT);    // 16 x 1024
    _Float16* Sbuf = (_Float16*)(smem + SM_SB);    // 16 x 2048
    float*    red  = (float*)   (smem + SM_RED);   // 16 x 16
    float*    rmax = (float*)   (smem + SM_RMAX);  // 16
    float*    rsum = (float*)   (smem + SM_RSUM);  // 16

    int tid  = threadIdx.x;
    int wave = tid >> 5;
    int lane = tid & 31;
    int half = lane >> 4;
    int lm   = lane & 15;
    int b    = blockIdx.x >> 7;          // 4 batches
    int q0   = (blockIdx.x & 127) << 4;  // 128 query blocks of 16

    // ---- Phase 0: stage Q block (16 x 1024 f16) in LDS ----
    const _Float16* qsrc = Qh + (size_t)(b * S_ + q0) * D_;
    for (int i = tid; i < (16 * D_) / 8; i += 256)
        *(v8h*)(Qt + i * 8) = *(const v8h*)(qsrc + i * 8);
    __syncthreads();

    // ---- Phase 1: scores for this wave's 256 keys (16 key-tile accs) ----
    const float scale = 0.125f;  // 1/sqrt(D_K=64)
    {
        const _Float16* arow = Qt + (size_t)lm * D_;
        const _Float16* krow = Kh + (size_t)(b * S_ + wave * 256 + lm) * D_ + half * 16;
        v8f acc[16];
#pragma unroll
        for (int t = 0; t < 16; ++t) acc[t] = (v8f){};

        v16h b0 = *(const v16h*)(krow + (size_t)0 * 16 * D_);
        v16h b1 = *(const v16h*)(krow + (size_t)1 * 16 * D_);
        v16h b2 = *(const v16h*)(krow + (size_t)2 * 16 * D_);
        v16h b3 = *(const v16h*)(krow + (size_t)3 * 16 * D_);

        for (int k0 = 0; k0 < D_; k0 += 32) {
            int k1 = (k0 + 32 < D_) ? (k0 + 32) : k0;
            v16h a = load_afrag(arow, k0, half);
#pragma unroll
            for (int t = 0; t < 16; ++t) {
                int tn = t + 4;
                int kn = (tn >= 16) ? k1 : k0;
                tn &= 15;
                v16h bnew = *(const v16h*)(krow + (size_t)tn * 16 * D_ + kn);
                acc[t] = wmma16x16x32(a, b0, acc[t]);
                b0 = b1; b1 = b2; b2 = b3; b3 = bnew;
            }
            SCHED_GROUP(0x100, 2, 0);                   // A-frag DS reads first
#pragma unroll
            for (int t = 0; t < 16; ++t) {              // (2 loads, 1 wmma) x16
                SCHED_GROUP(0x020, 2, 0);
                SCHED_GROUP(0x008, 1, 0);
            }
        }
#pragma unroll
        for (int t = 0; t < 16; ++t)
#pragma unroll
            for (int r = 0; r < 8; ++r)
                Sbuf[(r + 8 * half) * S_ + wave * 256 + t * 16 + lm] =
                    (_Float16)(acc[t][r] * scale);
    }
    __syncthreads();

    // ---- Phase 2: row softmax over 2048 keys (vectorized LDS I/O) ----
    {
        int row = tid & 15, sub = tid >> 4;          // 16 threads per row
        _Float16* srow = Sbuf + row * S_;
        int c0 = sub * 128;
        float mx = -3.402823466e38f;
        for (int c = 0; c < 128; c += 8) {
            v8h v = *(const v8h*)(srow + c0 + c);
#pragma unroll
            for (int e = 0; e < 8; ++e)
                mx = fmaxf(mx, (float)v[e]);
        }
        red[row * 16 + sub] = mx;
        __syncthreads();
        if (sub == 0) {
            float m = red[row * 16];
            for (int i = 1; i < 16; ++i) m = fmaxf(m, red[row * 16 + i]);
            rmax[row] = m;
        }
        __syncthreads();
        float M = rmax[row], s = 0.f;
        for (int c = 0; c < 128; c += 8) {
            v8h v = *(const v8h*)(srow + c0 + c);
            v8h p;
#pragma unroll
            for (int e = 0; e < 8; ++e) {
                float pe = __expf((float)v[e] - M);
                s += pe;
                p[e] = (_Float16)pe;
            }
            *(v8h*)(srow + c0 + c) = p;
        }
        red[row * 16 + sub] = s;
        __syncthreads();
        if (sub == 0) {
            float t2 = 0.f;
            for (int i = 1; i < 16; ++i) t2 += red[row * 16 + i];
            rsum[row] = t2 + red[row * 16];
        }
        __syncthreads();
    }

    // ---- Phase 3: O slice = P * V, normalize, write f32 ----
    {
        int n0 = wave * 128;                 // this wave's 128 output columns
        v8f acc[8] = {};
        const _Float16* prow  = Sbuf + (size_t)lm * S_;
        const _Float16* vrow  = XhT + ((size_t)b * D_ + n0 + lm) * S_ + half * 16;

        v16h a_cur = load_afrag(prow, 0, half);
        v16h b0 = *(const v16h*)(vrow + (size_t)0 * 16 * S_);
        v16h b1 = *(const v16h*)(vrow + (size_t)1 * 16 * S_);
        v16h b2 = *(const v16h*)(vrow + (size_t)2 * 16 * S_);
        v16h b3 = *(const v16h*)(vrow + (size_t)3 * 16 * S_);

        for (int k0 = 0; k0 < S_; k0 += 32) {
            int k1 = (k0 + 32 < S_) ? (k0 + 32) : k0;
            v16h a_nxt = load_afrag(prow, k1, half);
#pragma unroll
            for (int j = 0; j < 8; ++j) {
                int jn = j + 4;
                int kn = (jn >= 8) ? k1 : k0;
                jn &= 7;
                v16h bnew = *(const v16h*)(vrow + (size_t)jn * 16 * S_ + kn);
                acc[j] = wmma16x16x32(a_cur, b0, acc[j]);
                b0 = b1; b1 = b2; b2 = b3; b3 = bnew;
            }
            a_cur = a_nxt;
            SCHED_GROUP(0x100, 2, 0);                   // A-frag DS reads first
#pragma unroll
            for (int j = 0; j < 8; ++j) {               // (2 loads, 1 wmma) x8
                SCHED_GROUP(0x020, 2, 0);
                SCHED_GROUP(0x008, 1, 0);
            }
        }
#pragma unroll
        for (int r = 0; r < 8; ++r) {
            float inv = 1.0f / rsum[r + 8 * half];
            float* orow = out + (size_t)(b * S_ + q0 + r + 8 * half) * D_;
#pragma unroll
            for (int j = 0; j < 8; ++j)
                orow[n0 + j * 16 + lm] = acc[j][r] * inv;
        }
    }
}

// ---------------------------------------------------------------------------
// Host-side launcher.  Workspace layout (f16), 68 MB total:
//   Xh  [8M]  | XhT [8M] | WqT [1M] | WkT [1M] | Qh [8M] | Kh [8M]
// ---------------------------------------------------------------------------
extern "C" void kernel_launch(void* const* d_in, const int* in_sizes, int n_in,
                              void* d_out, int out_size, void* d_ws, size_t ws_size,
                              hipStream_t stream) {
    const float* X  = (const float*)d_in[0];
    const float* Wq = (const float*)d_in[1];
    const float* Wk = (const float*)d_in[2];
    float* out = (float*)d_out;

    const size_t NX = (size_t)B_ * S_ * D_;   // 8388608
    const size_t NW = (size_t)D_ * D_;        // 1048576

    _Float16* Xh  = (_Float16*)d_ws;
    _Float16* XhT = Xh  + NX;
    _Float16* WqT = XhT + NX;
    _Float16* WkT = WqT + NW;
    _Float16* Qh  = WkT + NW;
    _Float16* Kh  = Qh  + NX;

    cvt_x<<<dim3((unsigned)(NX / 256)), dim3(256), 0, stream>>>(X, Xh, XhT);
    cvt_w<<<dim3((unsigned)(NW / 256)), dim3(256), 0, stream>>>(Wq, Wk, WqT, WkT);

    // 512 M-tiles * 8 N128-tiles * 2 outputs = 8192 wave-tiles, 8 waves/block
    proj<<<dim3(1024), dim3(256), 0, stream>>>(Xh, WqT, WkT, Qh, Kh);

    // one 16-query block per workgroup: 4 * 128 = 512 blocks
    attn<<<dim3(B_ * (S_ / 16)), dim3(256), SMEM_BYTES, stream>>>(Qh, Kh, XhT, out);
}
